// GraphECG_42717744726364
// MI455X (gfx1250) — matrix operand order repaired and checked by reading, not compiled, not run on hardware
//
#include <hip/hip_runtime.h>
#include <cmath>

// ---------------- types ----------------
typedef _Float16 v16h __attribute__((ext_vector_type(16)));
typedef _Float16 v8h  __attribute__((ext_vector_type(8)));
typedef float    v8f  __attribute__((ext_vector_type(8)));

#define CAT16(a,b) __builtin_shufflevector(a,b,0,1,2,3,4,5,6,7,8,9,10,11,12,13,14,15)

#if __has_builtin(__builtin_amdgcn_tensor_load_to_lds) && __has_builtin(__builtin_amdgcn_s_wait_tensorcnt)
#define USE_TDM 1
typedef unsigned int u32x4 __attribute__((ext_vector_type(4)));
typedef int          i32x4 __attribute__((ext_vector_type(4)));
typedef int          i32x8 __attribute__((ext_vector_type(8)));
#endif

__device__ __forceinline__ int imin(int a,int b){return a<b?a:b;}
__device__ __forceinline__ int imax(int a,int b){return a>b?a:b;}

// single-instruction v_rcp_f32 (≈1 ulp) — adequate for activation math
__device__ __forceinline__ float rcpf(float x){
#if __has_builtin(__builtin_amdgcn_rcpf)
  return __builtin_amdgcn_rcpf(x);
#else
  return 1.f / x;
#endif
}

// mish(x) = x*tanh(softplus(x)) ; with u=e^x: tanh(log1p(u)) = (u^2+2u)/(u^2+2u+2)  (exact algebra)
// clamp exp arg: for x>=20 the ratio is 1.0 in f32, so the clamp is exact.
// Branchless: one v_exp_f32 + one v_rcp_f32 + a few FMAs.
__device__ __forceinline__ float mishf(float x){
  float u = __expf(fminf(x, 20.f));
  float n = u * (u + 2.f);
  return x * n * rcpf(n + 2.f);
}

// ---------------- problem constants ----------------
#define NE   1536
#define NT   5000
#define NN   1280
#define NB   128
#define L2D  1250   // final conv length
#define CHUNK 128
#define R1   288    // LDS rows for L1-domain buffers (need <= 275)
#define R2   144    // LDS rows for L2-domain buffers (need <= 134)

// packed f16 weight offsets (in halves)
#define OW1     0        // 64*32
#define OWR1A   2048     // 64*192
#define OWR1B   14336
#define OW2     26624    // 128*320
#define OWR2A   67584    // 128*384
#define OWR2B   116736
#define OW3     165888   // 192*384
#define WTOT    239616
// folded bias offsets (floats)
#define OB1    0
#define OBR1A  64
#define OBR1B  128
#define OB2    192
#define OBR2A  320
#define OBR2B  448
#define OB3    576

// LDS layout (halves)
#define S_IN    0
#define S_X1    (576)
#define S_R1M   (S_X1  + R1*64)
#define S_R1O   (S_R1M + R1*64)
#define S_X2    (S_R1O + R1*64)
#define S_R2M   (S_X2  + R2*128)
#define S_R2O   (S_R2M + R2*128)
#define S_HTOT  (S_R2O + R2*128)          // 111168 halves = 222336 B
#define S_STAGE_B (S_HTOT*2)              // f32[560] staging strip for TDM
#define S_ACC_B   (S_STAGE_B + 560*4)
#define ENC_SMEM  (S_ACC_B + 192*4)       // 225344 B total

// param leaf indices (setup_inputs insertion order, recursive)
#define P_NPW 5
#define P_NPB 6
#define P_GNN0 35
#define P_GPW 65
#define P_GPB 66
#define P_CLW 67
#define P_CLB 68

// ---------------- weight fold/convert ----------------
__global__ void fold_conv_kernel(const float* __restrict__ w, const float* __restrict__ cb,
                                 const float* __restrict__ g, const float* __restrict__ bb,
                                 int COUT,int CIN,int K,int KWC, _Float16* Wd, float* Bd){
  int i = blockIdx.x*blockDim.x + threadIdx.x;
  int total = COUT*KWC;
  const float inv = rsqrtf(1.f + 1e-5f);
  if (i < total){
    int co = i / KWC, kk = i % KWC;
    int dk = kk / CIN, ci = kk % CIN;
    float s = g[co] * inv;
    float v = (dk < K) ? w[(co*CIN + ci)*K + dk] * s : 0.f;
    Wd[(size_t)co*KWC + kk] = (_Float16)v;
  }
  if (i < COUT){
    float s = g[i] * inv;
    Bd[i] = cb[i]*s + bb[i];
  }
}

// ---------------- generic conv tile (implicit GEMM, WMMA f16->f32) ----------------
// kk = dk*CIN + ci ; activations in LDS as [pos][ci] ; MODE 0: mish->dst, 1: +skip,mish->dst, 2: mish->atomic acc
template<int COUT,int CIN,int KW,int PAD,int STRIDE,int MODE>
__device__ __forceinline__ void conv_tiles(const _Float16* __restrict__ W, const float* __restrict__ bias,
    const _Float16* sSrc, int src_lo,
    _Float16* sDst, int dst_lo,
    const _Float16* sSkip, int skip_lo,
    float* sAcc, int out_lo, int out_hi, int lane, int wave){
  constexpr int KWC = KW*CIN;          // multiple of 32 (CIN is 64 or 128)
  constexpr int NK  = KWC/32;
  const int Mt = COUT/16;
  const int Nt = (out_hi - out_lo + 15) >> 4;
  const int nn = lane & 15, hi = lane >> 4;
  for (int job = wave; job < Mt*Nt; job += 8){
    int mt = job / Nt, nt = job % Nt;
    int m0 = mt*16, n0 = out_lo + nt*16;
    int o  = n0 + nn;
    int oc = (o < out_hi) ? o : (out_hi - 1);
    v8f acc;
    #pragma unroll
    for (int j=0;j<8;++j) acc[j] = bias[m0 + 8*hi + j];
    #pragma unroll
    for (int ks=0; ks<NK; ++ks){
      const int kkb = ks*32;
      const int dk  = kkb / CIN;
      const int cib = kkb % CIN;
      const _Float16* wp = W + (size_t)(m0 + nn)*KWC + kkb + hi*8;
      v8h a0 = *(const v8h*)wp;
      v8h a1 = *(const v8h*)(wp + 16);
      v16h a = CAT16(a0, a1);
      int ia = STRIDE*oc + dk - PAD;                 // absolute src row
      const _Float16* bp = sSrc + (ia - src_lo)*CIN + cib + hi*8;
      v8h b0 = *(const v8h*)bp;
      v8h b1 = *(const v8h*)(bp + 16);
      v16h b = CAT16(b0, b1);
      acc = __builtin_amdgcn_wmma_f32_16x16x32_f16(false, a, false, b, (short)0, acc, false, false);
    }
    bool valid = (o < out_hi);
    if constexpr (MODE == 2){
      #pragma unroll
      for (int j=0;j<8;++j){
        float v = mishf(acc[j]);
        if (valid) atomicAdd(&sAcc[m0 + 8*hi + j], v);
      }
    } else {
      _Float16 ov[8] __attribute__((aligned(16)));
      #pragma unroll
      for (int j=0;j<8;++j){
        float v = acc[j];
        if constexpr (MODE == 1)
          v += (float)sSkip[(oc - skip_lo)*COUT + m0 + 8*hi + j];
        ov[j] = (_Float16)mishf(v);
      }
      if (valid) *(v8h*)(sDst + (oc - dst_lo)*COUT + m0 + 8*hi) = *(const v8h*)ov;
    }
  }
}

// c1: Cin=1, k=7 padded to K=32.  (i<7) is compile-time so loads are unconditional; hi-select is a cndmask.
__device__ __forceinline__ void conv1_tiles(const _Float16* __restrict__ W, const float* __restrict__ bias,
    const _Float16* sIn, int in_lo, _Float16* sDst, int dst_lo,
    int out_lo, int out_hi, int lane, int wave){
  const int Nt = (out_hi - out_lo + 15) >> 4;
  const int nn = lane & 15, hi = lane >> 4;
  for (int job = wave; job < 4*Nt; job += 8){
    int mt = job / Nt, nt = job % Nt;
    int m0 = mt*16, n0 = out_lo + nt*16;
    int o  = n0 + nn;
    int oc = (o < out_hi) ? o : (out_hi - 1);
    v8f acc;
    #pragma unroll
    for (int j=0;j<8;++j) acc[j] = bias[m0 + 8*hi + j];
    const _Float16* wp = W + (m0 + nn)*32 + hi*8;
    v8h a0 = *(const v8h*)wp;
    v8h a1 = *(const v8h*)(wp + 16);
    v16h a = CAT16(a0, a1);
    int base = 2*oc - 3 - in_lo;
    v16h b;
    #pragma unroll
    for (int i=0;i<16;++i){
      _Float16 v = (i < 7) ? sIn[base + i] : (_Float16)0.f;   // i compile-time
      b[i] = (hi == 0) ? v : (_Float16)0.f;                   // vector select, no branch
    }
    acc = __builtin_amdgcn_wmma_f32_16x16x32_f16(false, a, false, b, (short)0, acc, false, false);
    bool valid = (o < out_hi);
    _Float16 ov[8] __attribute__((aligned(16)));
    #pragma unroll
    for (int j=0;j<8;++j) ov[j] = (_Float16)mishf(acc[j]);
    if (valid) *(v8h*)(sDst + (oc - dst_lo)*64 + m0 + 8*hi) = *(const v8h*)ov;
  }
}

// ---------------- fused signal encoder: one workgroup per edge ----------------
__global__ __launch_bounds__(256) void enc_kernel(const float* __restrict__ sig,
    const _Float16* __restrict__ Wf, const float* __restrict__ Bf, float* __restrict__ eOut){
  extern __shared__ char smem[];
  _Float16* H    = (_Float16*)smem;
  _Float16* sIN  = H + S_IN;
  _Float16* sX1  = H + S_X1;
  _Float16* sR1M = H + S_R1M;
  _Float16* sR1O = H + S_R1O;
  _Float16* sX2  = H + S_X2;
  _Float16* sR2M = H + S_R2M;
  _Float16* sR2O = H + S_R2O;
  float* sSTG    = (float*)(smem + S_STAGE_B);
  float* sACC    = (float*)(smem + S_ACC_B);

  const int e = blockIdx.x;
  const int tid = threadIdx.x, lane = tid & 31, wave = tid >> 5;
  const float* in = sig + (size_t)e * NT;

  for (int t = tid; t < 192; t += 256) sACC[t] = 0.f;

  for (int ch = 0; ch < 10; ++ch){
    const int c0 = ch*CHUNK, c1 = imin(c0+CHUNK, L2D);
    __syncthreads();
    { // zero activation buffers + staging strip (not the mean accumulator)
      unsigned* z = (unsigned*)smem;
      const int nz = S_ACC_B/4;
      for (int t = tid; t < nz; t += 256) z[t] = 0u;
    }
    __syncthreads();
    // stage input chunk, zero-padded outside [0,NT)
    const int in_lo = 4*c0 - 23, in_hi = 4*c1 + 20;
    const int vlo = imax(in_lo, 0), vhi = imin(in_hi, NT);
    const int nval = vhi - vlo;
#ifdef USE_TDM
    // Tensor Data Mover: async DMA the valid f32 strip into LDS staging (wave 0 issues; EXEC-independent)
    if (wave == 0 && nval > 0){
      unsigned long long ga = (unsigned long long)(const void*)(in + vlo);
      u32x4 g0 = { 1u,                                             // count=1, user descriptor
                   (unsigned)(S_STAGE_B + (vlo - in_lo)*4),        // lds_addr (bytes)
                   (unsigned)(ga & 0xFFFFFFFFu),                   // global_addr[31:0]
                   (unsigned)((ga >> 32) & 0x01FFFFFFu) | (2u<<30) // global_addr[56:32] | type=2
                 };
      i32x8 g1 = { (int)(2u << 16),                                // data_size = 4 bytes
                   (int)((unsigned)(nval & 0xFFFF) << 16),         // tensor_dim0[15:0]
                   (int)((nval >> 16) & 0xFFFF),                   // tensor_dim0[31:16]
                   (int)((unsigned)(nval & 0xFFFF) << 16),         // tile_dim0
                   0, 0, 0, 0 };
      i32x4 z4 = { 0, 0, 0, 0 };
#if __clang_major__ >= 23
      i32x8 z8 = { 0, 0, 0, 0, 0, 0, 0, 0 };
      __builtin_amdgcn_tensor_load_to_lds(g0, g1, z4, z4, z8, 0);
#else
      __builtin_amdgcn_tensor_load_to_lds(g0, g1, z4, z4, 0);
#endif
      __builtin_amdgcn_s_wait_tensorcnt(0);
    }
    __syncthreads();
    for (int t = tid; t < in_hi - in_lo; t += 256)
      sIN[t] = (_Float16)sSTG[t];
#else
    for (int t = tid; t < in_hi - in_lo; t += 256){
      int a = in_lo + t;
      sIN[t] = (a >= 0 && a < NT) ? (_Float16)in[a] : (_Float16)0.f;
    }
#endif
    __syncthreads();
    // c1: input -> x1   (L1 domain rows [2c0-10, 2c1+9))
    const int x1_lo = 2*c0 - 10, x1_hi = 2*c1 + 9;
    conv1_tiles(Wf+OW1, Bf+OB1, sIN, in_lo, sX1, x1_lo, imax(x1_lo,0), imin(x1_hi,2500), lane, wave);
    __syncthreads();
    // rb1 conv a: x1 -> r1m
    const int r1m_lo = 2*c0 - 9, r1m_hi = 2*c1 + 8;
    conv_tiles<64,64,3,1,1,0>(Wf+OWR1A, Bf+OBR1A, sX1, x1_lo, sR1M, r1m_lo, nullptr, 0, nullptr,
                              imax(r1m_lo,0), imin(r1m_hi,2500), lane, wave);
    __syncthreads();
    // rb1 conv b + skip(x1): r1m -> r1o
    const int r1o_lo = 2*c0 - 8, r1o_hi = 2*c1 + 7;
    conv_tiles<64,64,3,1,1,1>(Wf+OWR1B, Bf+OBR1B, sR1M, r1m_lo, sR1O, r1o_lo, sX1, x1_lo, nullptr,
                              imax(r1o_lo,0), imin(r1o_hi,2500), lane, wave);
    __syncthreads();
    // c2 (stride 2): r1o -> x2
    const int x2_lo = c0 - 3, x2_hi = c1 + 3;
    conv_tiles<128,64,5,2,2,0>(Wf+OW2, Bf+OB2, sR1O, r1o_lo, sX2, x2_lo, nullptr, 0, nullptr,
                               imax(x2_lo,0), imin(x2_hi,L2D), lane, wave);
    __syncthreads();
    // rb2 conv a: x2 -> r2m
    const int r2m_lo = c0 - 2, r2m_hi = c1 + 2;
    conv_tiles<128,128,3,1,1,0>(Wf+OWR2A, Bf+OBR2A, sX2, x2_lo, sR2M, r2m_lo, nullptr, 0, nullptr,
                                imax(r2m_lo,0), imin(r2m_hi,L2D), lane, wave);
    __syncthreads();
    // rb2 conv b + skip(x2): r2m -> r2o
    const int r2o_lo = c0 - 1, r2o_hi = c1 + 1;
    conv_tiles<128,128,3,1,1,1>(Wf+OWR2B, Bf+OBR2B, sR2M, r2m_lo, sR2O, r2o_lo, sX2, x2_lo, nullptr,
                                imax(r2o_lo,0), imin(r2o_hi,L2D), lane, wave);
    __syncthreads();
    // c3 + mish + mean accumulate
    conv_tiles<192,128,3,1,1,2>(Wf+OW3, Bf+OB3, sR2O, r2o_lo, nullptr, 0, nullptr, 0, sACC,
                                c0, c1, lane, wave);
  }
  __syncthreads();
  for (int t = tid; t < 192; t += 256) eOut[(size_t)e*192 + t] = sACC[t] * (1.f/1250.f);
}

// ---------------- node features (sph_harm) + node_proj ----------------
__global__ __launch_bounds__(128) void node_kernel(const float* __restrict__ pos,
    const float* __restrict__ w, const float* __restrict__ b, float* __restrict__ xout){
  const int nidx = blockIdx.x, t = threadIdx.x;
  float X = pos[nidx*3+0], Y = pos[nidx*3+1], Z = pos[nidx*3+2];
  float r = fmaxf(sqrtf(X*X + Y*Y + Z*Z), 1e-8f);
  float ir = rcpf(r);
  float xn = X*ir, yn = Y*ir, zn = Z*ir;
  float x2 = xn*xn, y2 = yn*yn, z2 = zn*zn;
  float xy = xn*yn, xz = xn*zn, yz = yn*zn;
  const float c1 = 0.4886025119029199f, c2 = 1.0925484305920792f, c20 = 0.31539156525252005f;
  float f[29];
  f[0]=0.28209479177387814f; f[1]=c1*yn; f[2]=c1*zn; f[3]=c1*xn;
  f[4]=c2*xy; f[5]=c2*yz; f[6]=c20*(3.f*z2-1.f); f[7]=c2*xz; f[8]=0.5f*c2*(x2-y2);
  f[9]=yn*(3.f*x2-y2); f[10]=xn*yn*zn; f[11]=yn*(5.f*z2-1.f); f[12]=zn*(5.f*z2-3.f);
  f[13]=xn*(5.f*z2-1.f); f[14]=zn*(x2-y2); f[15]=xn*(x2-3.f*y2);
  f[16]=xy*(x2-y2); f[17]=yz*(3.f*x2-y2); f[18]=xy*(7.f*z2-1.f); f[19]=yz*(7.f*z2-3.f);
  f[20]=35.f*z2*z2-30.f*z2+3.f; f[21]=xz*(7.f*z2-3.f); f[22]=(x2-y2)*(7.f*z2-1.f);
  f[23]=xz*(x2-3.f*y2); f[24]=x2*x2-6.f*x2*y2+y2*y2;
  const float pi = 3.14159265358979323846f;
  f[25]=r; f[26]=sinf(pi*r); f[27]=cosf(pi*r); f[28]=expf(-r);
  float acc = b[t];
  #pragma unroll
  for (int k=0;k<29;++k) acc += w[t*29+k]*f[k];
  xout[nidx*128 + t] = acc;
}

// ---------------- GNN ----------------
__global__ void zero_kernel(float* p, int n){
  int i = blockIdx.x*blockDim.x + threadIdx.x;
  if (i < n) p[i] = 0.f;
}

__global__ __launch_bounds__(256) void gnn_msg_kernel(const float* __restrict__ x, const float* __restrict__ eattr,
    const int* __restrict__ src, const int* __restrict__ dst,
    const float* __restrict__ w1, const float* __restrict__ b1,
    const float* __restrict__ w2, const float* __restrict__ b2, float* __restrict__ agg){
  __shared__ float in[448];
  __shared__ float h[256];
  const int e = blockIdx.x, t = threadIdx.x;
  const int s = src[e], d = dst[e];
  if (t < 128) in[t] = x[(size_t)d*128 + t];
  else         in[t] = x[(size_t)s*128 + (t-128)];
  for (int k = t; k < 192; k += 256) in[256+k] = eattr[(size_t)e*192 + k];
  __syncthreads();
  float a = b1[t];
  for (int k=0;k<448;++k) a += w1[(size_t)t*448 + k]*in[k];
  h[t] = mishf(a);
  __syncthreads();
  float a2 = b2[t];
  for (int k=0;k<256;++k) a2 += w2[(size_t)t*256 + k]*h[k];
  atomicAdd(&agg[(size_t)d*256 + t], a2);
}

__global__ __launch_bounds__(256) void gnn_upd_kernel(const float* __restrict__ x, const float* __restrict__ agg,
    const float* __restrict__ w1, const float* __restrict__ b1,
    const float* __restrict__ w2, const float* __restrict__ b2,
    const float* __restrict__ g, const float* __restrict__ beta, float* __restrict__ xout){
  __shared__ float in[384];
  __shared__ float h[256];
  __shared__ float y[128];
  __shared__ float red[128];
  __shared__ float stat[2];
  const int nidx = blockIdx.x, t = threadIdx.x;
  if (t < 128) in[t] = x[(size_t)nidx*128 + t];
  in[128+t] = agg[(size_t)nidx*256 + t];
  __syncthreads();
  float a = b1[t];
  for (int k=0;k<384;++k) a += w1[(size_t)t*384 + k]*in[k];
  h[t] = mishf(a);
  __syncthreads();
  if (t < 128){
    float a2 = b2[t];
    for (int k=0;k<256;++k) a2 += w2[(size_t)t*256 + k]*h[k];
    float yy = in[t] + a2;
    y[t] = yy; red[t] = yy;
  }
  __syncthreads();
  for (int s2=64; s2>0; s2>>=1){ if (t < s2) red[t] += red[t+s2]; __syncthreads(); }
  if (t == 0) stat[0] = red[0] * (1.f/128.f);
  __syncthreads();
  const float mu = stat[0];
  if (t < 128){ float d = y[t]-mu; red[t] = d*d; }
  __syncthreads();
  for (int s2=64; s2>0; s2>>=1){ if (t < s2) red[t] += red[t+s2]; __syncthreads(); }
  if (t == 0) stat[1] = red[0] * (1.f/128.f);
  __syncthreads();
  if (t < 128)
    xout[(size_t)nidx*128 + t] = (y[t]-mu)*rsqrtf(stat[1] + 1e-5f)*g[t] + beta[t];
}

// ---------------- pooling + heads ----------------
__global__ void pool_accum_kernel(const float* __restrict__ x, const int* __restrict__ batch,
                                  float* __restrict__ gsum, float* __restrict__ cnt){
  int i = blockIdx.x*blockDim.x + threadIdx.x;
  if (i >= NN*128) return;
  int nidx = i >> 7, t = i & 127;
  int b = batch[nidx];
  atomicAdd(&gsum[(size_t)b*128 + t], x[i]);
  if (t == 0) atomicAdd(&cnt[b], 1.f);
}

__global__ __launch_bounds__(192) void head_kernel(const float* __restrict__ gsum, const float* __restrict__ cnt,
    const float* __restrict__ tab, const float* __restrict__ gpw, const float* __restrict__ gpb,
    const float* __restrict__ cw, const float* __restrict__ cb, float* __restrict__ out){
  __shared__ float ge[128];
  __shared__ float red[192];
  const int g = blockIdx.x, t = threadIdx.x;
  float c = fmaxf(cnt[g], 1.f);
  float ic = rcpf(c);
  if (t < 128) ge[t] = gsum[(size_t)g*128 + t] * ic;
  __syncthreads();
  float a = gpb[t];
  for (int k=0;k<128;++k) a += gpw[(size_t)t*128 + k]*ge[k];
  float v = mishf(a);
  out[NB + (size_t)g*192 + t] = v;     // graph_embed (second output)
  red[t] = cw[t]*v;
  __syncthreads();
  if (t < 64) red[t] = red[t] + red[t+64] + red[t+128];
  __syncthreads();
  for (int s2=32; s2>0; s2>>=1){ if (t < s2) red[t] += red[t+s2]; __syncthreads(); }
  if (t == 0){
    float s = red[0] + cb[0];
    for (int i=0;i<7;++i) s += cw[192+i]*tab[g*7 + i];
    out[g] = s;                         // logits (first output)
  }
}

// ---------------- launch ----------------
extern "C" void kernel_launch(void* const* d_in, const int* in_sizes, int n_in,
                              void* d_out, int out_size, void* d_ws, size_t ws_size,
                              hipStream_t stream){
  (void)in_sizes; (void)n_in; (void)out_size; (void)ws_size;
  const float* pos       = (const float*)d_in[0];
  const float* edge_attr = (const float*)d_in[1];
  const float* tabular   = (const float*)d_in[2];
  const int*   edge_idx  = (const int*)d_in[3];
  const int*   batch     = (const int*)d_in[4];
  auto F = [&](int i){ return (const float*)d_in[i]; };

  // workspace carve
  char* w = (char*)d_ws;
  _Float16* Wf  = (_Float16*)w; w += WTOT*2;         // 479232 B
  float* Bf     = (float*)w;    w += 768*4;
  float* eemb   = (float*)w;    w += (size_t)NE*192*4;
  float* x0     = (float*)w;    w += (size_t)NN*128*4;
  float* x1     = (float*)w;    w += (size_t)NN*128*4;
  float* agg    = (float*)w;    w += (size_t)NN*256*4;
  float* gsum   = (float*)w;    w += (size_t)NB*128*4;
  float* cnt    = (float*)w;    w += NB*4;

  auto fold = [&](int wi,int bi,int gi,int bbi,int CO,int CI,int K,int KWC,_Float16* wd,float* bd){
    int total = CO*KWC;
    fold_conv_kernel<<<(total+255)/256, 256, 0, stream>>>(F(wi),F(bi),F(gi),F(bbi),CO,CI,K,KWC,wd,bd);
  };
  fold( 7, 8, 9,10,  64,  1,7, 32, Wf+OW1,   Bf+OB1);    // c1  + bn1
  fold(11,12,13,14,  64, 64,3,192, Wf+OWR1A, Bf+OBR1A);  // rb1.c1 + rb1.bn1
  fold(15,16,17,18,  64, 64,3,192, Wf+OWR1B, Bf+OBR1B);  // rb1.c2 + rb1.bn2
  fold(19,20,21,22, 128, 64,5,320, Wf+OW2,   Bf+OB2);    // c2  + bn2
  fold(23,24,25,26, 128,128,3,384, Wf+OWR2A, Bf+OBR2A);  // rb2.c1 + rb2.bn1
  fold(27,28,29,30, 128,128,3,384, Wf+OWR2B, Bf+OBR2B);  // rb2.c2 + rb2.bn2
  fold(31,32,33,34, 192,128,3,384, Wf+OW3,   Bf+OB3);    // c3  + bn3

  node_kernel<<<NN, 128, 0, stream>>>(pos, F(P_NPW), F(P_NPB), x0);

  enc_kernel<<<NE, 256, ENC_SMEM, stream>>>(edge_attr, Wf, Bf, eemb);

  const int* src = edge_idx;
  const int* dst = edge_idx + NE;
  float* xc = x0; float* xn = x1;
  for (int l = 0; l < 3; ++l){
    int p = P_GNN0 + 10*l;
    zero_kernel<<<(NN*256+255)/256, 256, 0, stream>>>(agg, NN*256);
    gnn_msg_kernel<<<NE, 256, 0, stream>>>(xc, eemb, src, dst, F(p),F(p+1),F(p+2),F(p+3), agg);
    gnn_upd_kernel<<<NN, 256, 0, stream>>>(xc, agg, F(p+4),F(p+5),F(p+6),F(p+7),F(p+8),F(p+9), xn);
    float* t = xc; xc = xn; xn = t;
  }

  zero_kernel<<<((NB*128+NB)+255)/256, 256, 0, stream>>>(gsum, NB*128+NB); // gsum + cnt contiguous
  pool_accum_kernel<<<(NN*128+255)/256, 256, 0, stream>>>(xc, batch, gsum, cnt);
  head_kernel<<<NB, 192, 0, stream>>>(gsum, cnt, tabular, F(P_GPW),F(P_GPB),F(P_CLW),F(P_CLB), (float*)d_out);
}
